// Attention_223338300000
// MI455X (gfx1250) — matrix operand order repaired and checked by reading, not compiled, not run on hardware
//
#include <hip/hip_runtime.h>

// -------------------------------------------------------------------------
// Causal GQA prefill attention for MI455X (gfx1250, wave32, WMMA + TDM).
// fp32 in/out, bf16 v_wmma_f32_16x16x32_bf16 matmuls, f32 online softmax.
// One WG = 8 waves = one (batch, head, 128 query rows); one wave = 16 rows.
// K/V tiles double-buffered in LDS via tensor_load_to_lds DMA (TENSORcnt),
// overlapped with WMMA compute; f32->bf16 convert pass in LDS.
// -------------------------------------------------------------------------

typedef __attribute__((ext_vector_type(16))) __bf16        v16bf;
typedef __attribute__((ext_vector_type(8)))  float         v8f;
typedef __attribute__((ext_vector_type(4)))  unsigned int  v4u;
typedef __attribute__((ext_vector_type(8)))  int           v8i;
typedef __attribute__((ext_vector_type(4)))  int           v4i;

#define H_Q    16
#define H_KV   4
#define DHEAD  128
#define QTILE  128           // query rows per workgroup
#define KTILE  32            // keys per main-loop iteration
#define NWAVE  8
#define SCALE  0.08838834764831845f
#define NEGINF (-1.0e30f)

#if __has_builtin(__builtin_amdgcn_tensor_load_to_lds)
#define USE_TDM 1
#else
#define USE_TDM 0
#endif

__device__ __forceinline__ __bf16 f2bf(float f) {
  unsigned u = __float_as_uint(f);
  u += 0x7FFFu + ((u >> 16) & 1u);          // round-to-nearest-even
  unsigned short s = (unsigned short)(u >> 16);
  __bf16 r;
  __builtin_memcpy(&r, &s, sizeof(r));
  return r;
}

// ds_swizzle XOR butterfly within 32 lanes (masks <16 keep halves separate)
template <int M>
__device__ __forceinline__ float swz_xor(float x) {
  return __int_as_float(
      __builtin_amdgcn_ds_swizzle(__float_as_int(x), (M << 10) | 0x1F));
}

#if USE_TDM
// Issue a TDM 2D tile load: KTILE rows x DHEAD f32, row stride 512 elems.
__device__ __forceinline__ void tdm_load_tile(const float* gsrc,
                                              unsigned int lds_byte) {
  unsigned long long ga = (unsigned long long)(uintptr_t)gsrc;
  v4u g0;
  g0.x = 1u;                                            // count=1, load desc
  g0.y = lds_byte;                                      // lds_addr
  g0.z = (unsigned int)(ga & 0xFFFFFFFFu);              // global_addr[31:0]
  g0.w = (unsigned int)((ga >> 32) & 0x1FFFFFFu)        // global_addr[56:32]
         | (2u << 30);                                  // type = 2 (image)
  v8i g1;
  g1[0] = (2 << 16);                                    // data_size = 4B
  g1[1] = (DHEAD & 0xFFFF) << 16;                       // tensor_dim0 lo
  g1[2] = ((KTILE & 0xFFFF) << 16);                     // tdim0 hi=0|tdim1 lo
  g1[3] = ((DHEAD & 0xFFFF) << 16);                     // tdim1 hi=0|tile_dim0
  g1[4] = (KTILE & 0xFFFF);                             // tile_dim1, tile_dim2=0
  g1[5] = H_KV * DHEAD;                                 // dim0_stride = 512
  g1[6] = 0;
  g1[7] = 0;
  v4i z4 = {0, 0, 0, 0};
#if __has_include(<hip/amd_detail/amd_gfx1250_TDM.h>)
  v8i z8 = {0, 0, 0, 0, 0, 0, 0, 0};                    // clang-23 6-arg form
  __builtin_amdgcn_tensor_load_to_lds(g0, g1, z4, z4, z8, 0);
#else
  __builtin_amdgcn_tensor_load_to_lds(g0, g1, z4, z4, 0); // ROCm 7.2 5-arg
#endif
}
#endif

__global__ __launch_bounds__(256)
void Attention_223338300000_kernel(const float* __restrict__ q,
                                   const float* __restrict__ kin,
                                   const float* __restrict__ vin,
                                   const int*   __restrict__ slot_map,
                                   const int*   __restrict__ seqlen_ptr,
                                   float*       __restrict__ out)
{
#if USE_TDM
  __shared__ float rawK[2][KTILE * DHEAD];              // 2 x 16 KB (DMA dst)
  __shared__ float rawV[2][KTILE * DHEAD];              // 2 x 16 KB (DMA dst)
  __shared__ float validk[KTILE];
#endif
  __shared__ __align__(32) __bf16 Klds [KTILE * DHEAD];       // 8 KB [key][d]
  __shared__ __align__(32) __bf16 Vtlds[DHEAD * KTILE];       // 8 KB [d][key]
  __shared__ __align__(32) __bf16 Plds [NWAVE * 16 * KTILE];  // 8 KB bounce

  const int S    = *seqlen_ptr;
  const int tid  = threadIdx.x;
  const int wave = tid >> 5;
  const int lane = tid & 31;
  const int n16  = lane & 15;
  const int half = lane >> 4;

  const int h    = blockIdx.x % H_Q;
  const int rt   = blockIdx.x / H_Q;
  const int row0 = rt * QTILE;            // packed-token row of WG tile
  const int b    = row0 / S;
  const int q0wg = row0 % S;              // sequence-local first row of WG
  const int q0   = q0wg + wave * 16;      // sequence-local first row of wave
  const int kvh  = h / (H_Q / H_KV);

  // ---- Q fragments in WMMA A-layout; softmax scale folded into Q ----
  v16bf qa[4];
  {
    const size_t rowbase =
        (size_t)(b * S + q0 + n16) * (H_Q * DHEAD) + (size_t)h * DHEAD;
    const float* qrow = q + rowbase;
#pragma unroll
    for (int c = 0; c < 4; ++c) {
#pragma unroll
      for (int e = 0; e < 8; ++e) {
        qa[c][e]     = f2bf(qrow[c * 32 + half * 16 +     e] * SCALE);
        qa[c][8 + e] = f2bf(qrow[c * 32 + half * 16 + 8 + e] * SCALE);
      }
    }
  }

  v8f acc[8] = {};                        // 16 x 128 f32 output accumulator
  float mrow[8], rsum[8];
#pragma unroll
  for (int r = 0; r < 8; ++r) { mrow[r] = NEGINF; rsum[r] = 0.0f; }

  const int nkt = (q0wg + QTILE) / KTILE; // causal bound for whole WG
  const size_t kvstride = (size_t)(H_KV * DHEAD);

#if USE_TDM
  // Prologue: DMA tile 0 into buffer 0 (wave 0 owns the TDM engine).
  if (wave == 0) {
    const size_t gb = (size_t)(b * S) * kvstride + (size_t)kvh * DHEAD;
    tdm_load_tile(kin + gb, (unsigned int)(uintptr_t)&rawK[0][0]);
    tdm_load_tile(vin + gb, (unsigned int)(uintptr_t)&rawV[0][0]);
  }
#endif

  for (int kt = 0; kt < nkt; ++kt) {
    const int kbase = kt * KTILE;

#if USE_TDM
    const int buf = kt & 1;
    if (wave == 0) __builtin_amdgcn_s_wait_tensorcnt(0);  // DMA(kt) complete
    if (tid < KTILE)
      validk[tid] = (slot_map[b * S + kbase + tid] >= 0) ? 1.0f : 0.0f;
    __syncthreads();   // also: previous compute done reading Klds/Vtlds

    // Overlap: issue DMA for tile kt+1 into the other buffer.
    if ((wave == 0) && (kt + 1 < nkt)) {
      const size_t gb =
          (size_t)(b * S + kbase + KTILE) * kvstride + (size_t)kvh * DHEAD;
      tdm_load_tile(kin + gb, (unsigned int)(uintptr_t)&rawK[buf ^ 1][0]);
      tdm_load_tile(vin + gb, (unsigned int)(uintptr_t)&rawV[buf ^ 1][0]);
    }

    // Convert pass: raw f32 tiles -> bf16 K (row major) / V (transposed).
#pragma unroll
    for (int i = 0; i < (KTILE * DHEAD) / 256; ++i) {
      const int idx = tid + i * 256;
      const int key = idx >> 7;
      const int d   = idx & (DHEAD - 1);
      const float g = validk[key];        // invalid slot => zero row
      Klds[key * DHEAD + d]  = f2bf(rawK[buf][idx] * g);
      Vtlds[d * KTILE + key] = f2bf(rawV[buf][idx] * g);
    }
    __syncthreads();
#else
    __syncthreads();
#pragma unroll
    for (int i = 0; i < (KTILE * DHEAD) / 256; ++i) {
      const int idx = tid + i * 256;
      const int key = idx >> 7;
      const int d   = idx & (DHEAD - 1);
      const int tok = b * S + kbase + key;
      const bool ok = slot_map[tok] >= 0;
      const size_t g = (size_t)tok * kvstride + (size_t)kvh * DHEAD + d;
      Klds[key * DHEAD + d]  = f2bf(ok ? kin[g] : 0.0f);
      Vtlds[d * KTILE + key] = f2bf(ok ? vin[g] : 0.0f);
    }
    __syncthreads();
#endif

    if (kbase > q0 + 15) continue;        // tile fully masked for this wave

    // ---- scores = (Q*scale) * K^T : 8 WMMAs ----
    v8f s0 = {}, s1 = {};
#pragma unroll
    for (int c = 0; c < 4; ++c) {
      const v16bf bk0 =
          *(const v16bf*)&Klds[(n16)      * DHEAD + c * 32 + half * 16];
      s0 = __builtin_amdgcn_wmma_f32_16x16x32_bf16(
               false, qa[c], false, bk0, (short)0, s0, false, false);
      const v16bf bk1 =
          *(const v16bf*)&Klds[(16 + n16) * DHEAD + c * 32 + half * 16];
      s1 = __builtin_amdgcn_wmma_f32_16x16x32_bf16(
               false, qa[c], false, bk1, (short)0, s1, false, false);
    }

    // ---- causal mask + online softmax (ds_swizzle XOR butterflies) ----
#pragma unroll
    for (int r = 0; r < 8; ++r) {
      const int mglob = q0 + r + 8 * half;          // sequence-local row
      const float x0 = (kbase + n16      <= mglob) ? s0[r] : NEGINF;
      const float x1 = (kbase + 16 + n16 <= mglob) ? s1[r] : NEGINF;

      float mx = fmaxf(x0, x1);
      mx = fmaxf(mx, swz_xor<1>(mx));
      mx = fmaxf(mx, swz_xor<2>(mx));
      mx = fmaxf(mx, swz_xor<4>(mx));
      mx = fmaxf(mx, swz_xor<8>(mx));

      const float mnew  = fmaxf(mrow[r], mx);
      const float alpha = __expf(mrow[r] - mnew);
      const float msub  = (mnew <= -1.0e29f) ? 0.0f : mnew;
      const float p0 = __expf(x0 - msub);
      const float p1 = __expf(x1 - msub);

      float ps = p0 + p1;
      ps += swz_xor<1>(ps);
      ps += swz_xor<2>(ps);
      ps += swz_xor<4>(ps);
      ps += swz_xor<8>(ps);

      rsum[r] = rsum[r] * alpha + ps;
      mrow[r] = mnew;
#pragma unroll
      for (int dc = 0; dc < 8; ++dc) acc[dc][r] *= alpha;

      // P (C-layout) -> per-wave LDS bounce, bf16
      const int m = r + 8 * half;
      Plds[wave * (16 * KTILE) + m * KTILE + n16]      = f2bf(p0);
      Plds[wave * (16 * KTILE) + m * KTILE + 16 + n16] = f2bf(p1);
    }

    // wave-local LDS visibility (no WG barrier: other waves untouched)
    asm volatile("s_wait_dscnt 0" ::: "memory");

    // ---- reload P in WMMA A-layout ----
    union { v16bf v; uint4 u[2]; } pa;
    {
      const __bf16* pb = &Plds[wave * (16 * KTILE) + n16 * KTILE];
      pa.u[0] = *(const uint4*)(pb + half * 16);
      pa.u[1] = *(const uint4*)(pb + half * 16 + 8);
    }

    // ---- O += P * V : 8 WMMAs over d-chunks ----
#pragma unroll
    for (int dc = 0; dc < 8; ++dc) {
      const v16bf bv =
          *(const v16bf*)&Vtlds[(dc * 16 + n16) * KTILE + half * 16];
      acc[dc] = __builtin_amdgcn_wmma_f32_16x16x32_bf16(
                    false, pa.v, false, bv, (short)0, acc[dc], false, false);
    }
  }

  // ---- normalize and store ----
#pragma unroll
  for (int r = 0; r < 8; ++r) {
    const int m = r + 8 * half;
    const float inv = 1.0f / rsum[r];
    const size_t obase =
        (size_t)(b * S + q0 + m) * (H_Q * DHEAD) + (size_t)h * DHEAD;
#pragma unroll
    for (int dc = 0; dc < 8; ++dc) {
      out[obase + dc * 16 + n16] = acc[dc][r] * inv;
    }
  }
}

extern "C" void kernel_launch(void* const* d_in, const int* in_sizes, int n_in,
                              void* d_out, int out_size, void* d_ws, size_t ws_size,
                              hipStream_t stream) {
  // inputs: q, k, v, k_cache, v_cache, slot_mapping, batch, seqlen
  const float* q    = (const float*)d_in[0];
  const float* k    = (const float*)d_in[1];
  const float* v    = (const float*)d_in[2];
  const int*   slot = (const int*)  d_in[5];
  const int*   seql = (const int*)  d_in[7];
  float*       out  = (float*)d_out;

  const int N = in_sizes[5];               // total packed tokens
  dim3 grid((N / QTILE) * H_Q);
  dim3 block(256);
  Attention_223338300000_kernel<<<grid, block, 0, stream>>>(q, k, v, slot,
                                                            seql, out);
}
